// BinaryLinear_14259291422933
// MI455X (gfx1250) — matrix operand order repaired and checked by reading, not compiled, not run on hardware
//
#include <hip/hip_runtime.h>

typedef __attribute__((ext_vector_type(16))) _Float16 v16h;
typedef __attribute__((ext_vector_type(8)))  float    v8f;

#define IN_F   1024
#define OUT_F  1024
#define BATCHN 32768
#define EPSV   1e-6f

// ---------------------------------------------------------------------------
// Prep: per-output-row scale = max(mean|W[o,:]|, eps) and transposed binarized
// weights Wt[k][o] = sign(W[o][k]) in f16 (exactly representable).
// One wave per output row; uniform control flow (1024 waves exactly).
// ---------------------------------------------------------------------------
__global__ __launch_bounds__(256)
void bl_prep_kernel(const float* __restrict__ W,
                    _Float16* __restrict__ Wt,     // [IN_F][OUT_F]
                    float* __restrict__ scale) {   // [OUT_F]
  const int o    = (blockIdx.x * blockDim.x + threadIdx.x) >> 5;
  const int lane = threadIdx.x & 31;
  const float* wrow = W + (size_t)o * IN_F;
  float s = 0.f;
#pragma unroll 4
  for (int k = lane; k < IN_F; k += 32) {
    const float w = wrow[k];
    s += fabsf(w);
    Wt[(size_t)k * OUT_F + o] = (w >= 0.f) ? (_Float16)1.0f : (_Float16)(-1.0f);
  }
#pragma unroll
  for (int off = 16; off > 0; off >>= 1)
    s += __shfl_xor(s, off, 32);
  if (lane == 0)
    scale[o] = fmaxf(s * (1.0f / (float)IN_F), EPSV);
}

// ---------------------------------------------------------------------------
// GEMM: out[m][n] = scale[n] * sum_k x[m][k]*Wt[k][n] + bias[n]
// Block: 256 threads = 8 waves, block tile 128(M) x 128(N), K step 32.
// Wave tile: 64(M) x 32(N) = 4x2 WMMA 16x16 accumulators.
// ---------------------------------------------------------------------------
__global__ __launch_bounds__(256)
void bl_gemm_kernel(const float* __restrict__ x,
                    const _Float16* __restrict__ Wt,
                    const float* __restrict__ scale,
                    const float* __restrict__ bias,
                    float* __restrict__ out) {
  const int lane = threadIdx.x & 31;
  const int wave = threadIdx.x >> 5;
  const int wm = wave & 1;        // 0..1 : M sub-block
  const int wn = wave >> 1;       // 0..3 : N sub-block
  const int m0 = blockIdx.y * 128 + wm * 64;
  const int n0 = blockIdx.x * 128 + wn * 32;
  const int l15 = lane & 15;
  const int lhi = lane >> 4;      // 0/1

  // A fragment base: lane holds row m0+mt*16+l15; halves 0-7 = K(+8*lhi)+0..7,
  // halves 8-15 = same +16  (ISA 16-bit A 16x32 layout).
  const float* arow0 = x + (size_t)(m0 + l15) * IN_F + 8 * lhi;

  // B fragment base: lane's N = l15; halves 0-7 = K(+8*lhi)+0..7, halves 8-15 = same +16.
  const _Float16* bbase = Wt + (size_t)(8 * lhi) * OUT_F + n0 + l15;

  v8f acc[4][2];
#pragma unroll
  for (int mt = 0; mt < 4; ++mt)
#pragma unroll
    for (int nt = 0; nt < 2; ++nt)
      acc[mt][nt] = (v8f){};

  for (int k0 = 0; k0 < IN_F; k0 += 32) {
    v16h a[4];
#pragma unroll
    for (int mt = 0; mt < 4; ++mt) {
      const float* p = arow0 + (size_t)mt * 16 * IN_F + k0;
      const float4 f0 = *(const float4*)(p + 0);
      const float4 f1 = *(const float4*)(p + 4);
      const float4 f2 = *(const float4*)(p + 16);
      const float4 f3 = *(const float4*)(p + 20);
      v16h av = { (_Float16)f0.x, (_Float16)f0.y, (_Float16)f0.z, (_Float16)f0.w,
                  (_Float16)f1.x, (_Float16)f1.y, (_Float16)f1.z, (_Float16)f1.w,
                  (_Float16)f2.x, (_Float16)f2.y, (_Float16)f2.z, (_Float16)f2.w,
                  (_Float16)f3.x, (_Float16)f3.y, (_Float16)f3.z, (_Float16)f3.w };
      a[mt] = av;
    }

    v16h b[2];
#pragma unroll
    for (int nt = 0; nt < 2; ++nt) {
      const _Float16* q = bbase + (size_t)k0 * OUT_F + nt * 16;
#pragma unroll
      for (int s = 0; s < 16; ++s)
        b[nt][s] = q[(size_t)((s & 7) + ((s >> 3) << 4)) * OUT_F];
    }

#pragma unroll
    for (int mt = 0; mt < 4; ++mt)
#pragma unroll
      for (int nt = 0; nt < 2; ++nt)
        acc[mt][nt] = __builtin_amdgcn_wmma_f32_16x16x32_f16(
            false, a[mt], false, b[nt], (short)0, acc[mt][nt], false, false);
  }

  // Epilogue: C/D layout -> VGPR r holds M = r + 8*lhi, N = l15.
  // Non-temporal stores keep the 128MB output stream from evicting x in L2.
#pragma unroll
  for (int nt = 0; nt < 2; ++nt) {
    const int col = n0 + nt * 16 + l15;
    const float sc = scale[col];
    const float bs = bias[col];
#pragma unroll
    for (int mt = 0; mt < 4; ++mt) {
      const int rowbase = m0 + mt * 16 + 8 * lhi;
#pragma unroll
      for (int r = 0; r < 8; ++r) {
        const float v = acc[mt][nt][r] * sc + bs;
        __builtin_nontemporal_store(v, out + (size_t)(rowbase + r) * OUT_F + col);
      }
    }
  }
}

extern "C" void kernel_launch(void* const* d_in, const int* in_sizes, int n_in,
                              void* d_out, int out_size, void* d_ws, size_t ws_size,
                              hipStream_t stream) {
  (void)in_sizes; (void)n_in; (void)out_size; (void)ws_size;
  const float* x = (const float*)d_in[0];
  const float* W = (const float*)d_in[1];
  const float* b = (const float*)d_in[2];
  float* out = (float*)d_out;

  // Workspace: 2MB transposed binary weights (f16) + 4KB scales.
  _Float16* Wt   = (_Float16*)d_ws;
  float*    scal = (float*)((char*)d_ws + (size_t)IN_F * OUT_F * sizeof(_Float16));

  bl_prep_kernel<<<dim3((OUT_F * 32) / 256), dim3(256), 0, stream>>>(W, Wt, scal);

  dim3 grid(OUT_F / 128, BATCHN / 128);   // N fastest -> x block reused across N-blocks in L2
  bl_gemm_kernel<<<grid, dim3(256), 0, stream>>>(x, Wt, scal, b, out);
}
